// GraphSAGEEncoder_48318382080414
// MI455X (gfx1250) — matrix operand order repaired and checked by reading, not compiled, not run on hardware
//
#include <hip/hip_runtime.h>

// ---------- types for WMMA ----------
typedef __attribute__((ext_vector_type(16))) __bf16 v16bf;
typedef __attribute__((ext_vector_type(2)))  __bf16 v2bf;
typedef __attribute__((ext_vector_type(8)))  float  v8f;

union Frag16 {
    unsigned int u[8];
    v16bf v;
};

#if defined(__has_builtin)
#if __has_builtin(__builtin_amdgcn_cvt_pk_bf16_f32)
#define HAVE_PK_BF16 1
#endif
#endif

__device__ __forceinline__ unsigned int pack_bf2(float lo, float hi) {
#ifdef HAVE_PK_BF16
    union { v2bf v; unsigned int u; } cv;
    cv.v = __builtin_amdgcn_cvt_pk_bf16_f32(lo, hi);
    return cv.u;
#else
    // native fptrunc: backend picks HW bf16 cvt if present, else RNE sequence
    union { v2bf v; unsigned int u; } cv;
    cv.v.x = (__bf16)lo;
    cv.v.y = (__bf16)hi;
    return cv.u;
#endif
}

// ---------- utility kernels ----------
__global__ void zero_kernel(float* __restrict__ p, size_t n) {
    size_t i = (size_t)blockIdx.x * blockDim.x + threadIdx.x;
    size_t stride = (size_t)gridDim.x * blockDim.x;
    for (; i < n; i += stride) p[i] = 0.0f;
}

__global__ void degree_kernel(const long long* __restrict__ dst,
                              float* __restrict__ deg, int nE) {
    int i = blockIdx.x * blockDim.x + threadIdx.x;
    int stride = gridDim.x * blockDim.x;
    for (; i < nE; i += stride) atomicAdd(&deg[(int)dst[i]], 1.0f);
}

__global__ void rdeg_kernel(float* __restrict__ deg, int n) {
    int i = blockIdx.x * blockDim.x + threadIdx.x;
    if (i < n) deg[i] = 1.0f / fmaxf(deg[i], 1.0f);
}

// One wave handles 32 edges per outer step: indices loaded once per wave
// (coalesced), broadcast with __shfl; feature rows gathered as float4.
// Requires D % 128 == 0 (holds: D = 128 or 256).
__global__ __launch_bounds__(256)
void scatter_kernel(const float* __restrict__ feat,
                    const long long* __restrict__ src,
                    const long long* __restrict__ dst,
                    float* __restrict__ msg, int nE, int D) {
    int w    = (blockIdx.x * blockDim.x + threadIdx.x) >> 5;
    int lane = threadIdx.x & 31;
    int nW   = (gridDim.x * blockDim.x) >> 5;
    for (int base = w * 32; base < nE; base += nW * 32) {
        int e = base + lane;
        int sv = 0, dv = 0;
        if (e < nE) {
            sv = (int)src[e];
            dv = (int)dst[e];
        }
        int cnt = min(32, nE - base);
        for (int j = 0; j < cnt; ++j) {
            int s = __shfl(sv, j);
            int d = __shfl(dv, j);
            const float* fs = feat + (size_t)s * D;
            float*       md = msg  + (size_t)d * D;
            for (int c = lane * 4; c < D; c += 128) {
                float4 v = *(const float4*)(fs + c);
                atomicAdd(&md[c + 0], v.x);
                atomicAdd(&md[c + 1], v.y);
                atomicAdd(&md[c + 2], v.z);
                atomicAdd(&md[c + 3], v.w);
            }
        }
    }
}

// ---------- fused SAGE GEMM ----------
// out[n, j] = act( (msg[n,:]*rdeg[n]) @ B0 + dense[n,:] @ B1 + bias )
// Two K phases (compile-time dims -> immediate offsets, exact trip counts):
//   phase 1: A=msg (scaled by rdeg),  B=B0
//   phase 2: A=dense (unscaled),      B=B1
// Block: 256 threads = 8 waves; tile = 128 rows x 32 cols; each wave owns a
// 16x32 slab -> 2x v_wmma_f32_16x16x32_bf16 per K-step. Global loads for the
// next K-step are issued before the WMMA section (register double-buffer).
#define LDA 36  // bf16 elems per LDS row: 72B rows -> 8B aligned, conflict-free
#define LDB 36

template <int KHALF, int DOUT, bool SCALE>
__device__ __forceinline__ void gemm_phase(
    const float* __restrict__ Asrc, const float* __restrict__ Bsrc,
    const float* __restrict__ rdeg,
    unsigned short* As, unsigned short* Bs,
    int N, int rowBase, int colBase,
    int tid, int wave, int lane, v8f& acc0, v8f& acc1) {
    // ---- K-invariant staging state ----
    const float* aptr[4];
    float        ascale[4];
    uint2*       asdst[4];
#pragma unroll
    for (int i = 0; i < 4; ++i) {
        int idx = i * 256 + tid;            // 0..1023 float4 slots
        int r   = idx >> 3;                 // 0..127
        int k4  = (idx & 7) * 4;            // 0,4,...,28
        int n   = min(rowBase + r, N - 1);  // clamped: junk rows never stored
        aptr[i]   = Asrc + (size_t)n * KHALF + k4;
        ascale[i] = SCALE ? rdeg[n] : 1.0f;
        asdst[i]  = (uint2*)&As[r * LDA + k4];
    }
    // B: one column x 4 consecutive K per thread (coalesced loads, b64 store)
    const int bc  = tid & 31;
    const int bk4 = (tid >> 5) * 4;
    const float* bptr = Bsrc + (size_t)bk4 * DOUT + colBase + bc;
    uint2* bsdst = (uint2*)&Bs[bc * LDB + bk4];

    // fragment source pointers (A and B element s holds k = 16*(lane/16) + s, contiguous)
    const int m  = lane & 15;
    const int kb = (lane < 16) ? 0 : 16;
    const unsigned short* arow = &As[(wave * 16 + m) * LDA];
    const int kbB = (lane < 16) ? 0 : 16;
    const unsigned short* brow0 = &Bs[m * LDB + kbB];
    const unsigned short* brow1 = &Bs[(16 + m) * LDB + kbB];

    float4 av[4];
    float4 bv;
    auto load_tiles = [&](int kk) {
#pragma unroll
        for (int i = 0; i < 4; ++i) av[i] = *(const float4*)(aptr[i] + kk);
        const float* bp = bptr + (size_t)kk * DOUT;
        bv.x = bp[0];
        bv.y = bp[DOUT];
        bv.z = bp[2 * DOUT];
        bv.w = bp[3 * DOUT];
    };

    load_tiles(0);
#pragma unroll
    for (int kk = 0; kk < KHALF; kk += 32) {
        __syncthreads();
        // --- stage current tiles to LDS (packed bf16) ---
#pragma unroll
        for (int i = 0; i < 4; ++i) {
            float4 v = av[i];
            if (SCALE) {
                v.x *= ascale[i]; v.y *= ascale[i];
                v.z *= ascale[i]; v.w *= ascale[i];
            }
            uint2 p;
            p.x = pack_bf2(v.x, v.y);
            p.y = pack_bf2(v.z, v.w);
            *asdst[i] = p;
        }
        {
            uint2 p;
            p.x = pack_bf2(bv.x, bv.y);
            p.y = pack_bf2(bv.z, bv.w);
            *bsdst = p;
        }
        __syncthreads();
        // --- prefetch next K-step while WMMAs run ---
        if (kk + 32 < KHALF) load_tiles(kk + 32);

        // --- fragments + WMMA ---
        Frag16 a, b0, b1;
#pragma unroll
        for (int p = 0; p < 4; ++p) {
            a.u[p]     = *(const unsigned int*)(arow + kb + 2 * p);
            a.u[4 + p] = *(const unsigned int*)(arow + kb + 8 + 2 * p);
        }
#pragma unroll
        for (int p = 0; p < 8; ++p) {
            b0.u[p] = *(const unsigned int*)(brow0 + 2 * p);
            b1.u[p] = *(const unsigned int*)(brow1 + 2 * p);
        }

        acc0 = __builtin_amdgcn_wmma_f32_16x16x32_bf16(
            false, a.v, false, b0.v, (short)0, acc0, false, false);
        acc1 = __builtin_amdgcn_wmma_f32_16x16x32_bf16(
            false, a.v, false, b1.v, (short)0, acc1, false, false);
    }
}

template <int KHALF, int DOUT, bool FUSE_BN>
__global__ __launch_bounds__(256)
void sage_gemm(const float* __restrict__ A0,    // msgsum  [N, KHALF]
               const float* __restrict__ A1,    // dense   [N, KHALF]
               const float* __restrict__ rdeg,  // [N]  (1/max(deg,1))
               const float* __restrict__ B0,    // [KHALF, DOUT]
               const float* __restrict__ B1,    // [KHALF, DOUT]
               const float* __restrict__ bias,  // [DOUT]
               const float* __restrict__ g,     // bn gamma
               const float* __restrict__ be,    // bn beta
               const float* __restrict__ mu,    // bn mean
               const float* __restrict__ var,   // bn var
               float* __restrict__ out,         // [N, DOUT]
               int N) {
    __shared__ __align__(16) unsigned short As[128 * LDA];  // 128 rows x 32 K
    __shared__ __align__(16) unsigned short Bs[32 * LDB];   // 32 cols  x 32 K

    const int tid  = threadIdx.x;
    const int wave = tid >> 5;
    const int lane = tid & 31;
    const int rowBase = blockIdx.y * 128;
    const int colBase = blockIdx.x * 32;

    v8f acc0 = {};
    v8f acc1 = {};

    gemm_phase<KHALF, DOUT, true >(A0, B0, rdeg, As, Bs, N, rowBase, colBase,
                                   tid, wave, lane, acc0, acc1);
    gemm_phase<KHALF, DOUT, false>(A1, B1, rdeg, As, Bs, N, rowBase, colBase,
                                   tid, wave, lane, acc0, acc1);

    // --- epilogue: per-column affine hoisted out of row loop ---
    // BN case: v = max(acc*s + o, 0), s = gamma*rsqrt(var+eps),
    //          o = (bias - mean)*s + beta.  Plain: v = acc + bias.
    const int m0 = (lane >> 4) * 8;
    const int c0 = lane & 15;
    const int colA = colBase + c0;
    const int colB = colBase + 16 + c0;
    float sA = 1.0f, oA, sB = 1.0f, oB;
    if (FUSE_BN) {
        sA = g[colA] * rsqrtf(var[colA] + 1e-5f);
        oA = (bias[colA] - mu[colA]) * sA + be[colA];
        sB = g[colB] * rsqrtf(var[colB] + 1e-5f);
        oB = (bias[colB] - mu[colB]) * sB + be[colB];
    } else {
        oA = bias[colA];
        oB = bias[colB];
    }
    float* orow = out + (size_t)(rowBase + wave * 16 + m0) * DOUT;
    const bool fullTile = (rowBase + 128 <= N);
    if (fullTile) {
#pragma unroll
        for (int r = 0; r < 8; ++r) {
            float vA = acc0[r] * sA + oA;
            float vB = acc1[r] * sB + oB;
            if (FUSE_BN) { vA = fmaxf(vA, 0.0f); vB = fmaxf(vB, 0.0f); }
            orow[(size_t)r * DOUT + colA] = vA;
            orow[(size_t)r * DOUT + colB] = vB;
        }
    } else {
#pragma unroll
        for (int r = 0; r < 8; ++r) {
            int row = rowBase + wave * 16 + m0 + r;
            if (row < N) {
                float vA = acc0[r] * sA + oA;
                float vB = acc1[r] * sB + oB;
                if (FUSE_BN) { vA = fmaxf(vA, 0.0f); vB = fmaxf(vB, 0.0f); }
                out[(size_t)row * DOUT + colA] = vA;
                out[(size_t)row * DOUT + colB] = vB;
            }
        }
    }
}

// ---------- launch ----------
extern "C" void kernel_launch(void* const* d_in, const int* in_sizes, int n_in,
                              void* d_out, int out_size, void* d_ws, size_t ws_size,
                              hipStream_t stream) {
    const int D_IN = 128, D_HID = 256, D_OUT = 128;
    const float* x    = (const float*)d_in[0];
    const long long* edges = (const long long*)d_in[1];
    const float* W1l  = (const float*)d_in[2];
    const float* W1r  = (const float*)d_in[3];
    const float* b1   = (const float*)d_in[4];
    const float* bn_g = (const float*)d_in[5];
    const float* bn_b = (const float*)d_in[6];
    const float* bn_m = (const float*)d_in[7];
    const float* bn_v = (const float*)d_in[8];
    const float* W2l  = (const float*)d_in[9];
    const float* W2r  = (const float*)d_in[10];
    const float* b2   = (const float*)d_in[11];
    float* out = (float*)d_out;

    const int N  = in_sizes[0] / D_IN;
    const int nE = in_sizes[1] / 2;
    const long long* src = edges;
    const long long* dst = edges + nE;

    // workspace layout (floats)
    float* ws   = (float*)d_ws;
    float* deg  = ws;                          // [N] -> reciprocal after finalize
    float* msg1 = deg + N;                     // [N,128]
    float* h    = msg1 + (size_t)N * D_IN;     // [N,256]
    float* msg2 = h + (size_t)N * D_HID;       // [N,256]

    // 1) zero deg+msg1 and msg2
    {
        size_t n1 = (size_t)N * (1 + D_IN);
        size_t n2 = (size_t)N * D_HID;
        zero_kernel<<<(int)((n1 + 1023) / 1024), 256, 0, stream>>>(deg, n1);
        zero_kernel<<<(int)((n2 + 1023) / 1024), 256, 0, stream>>>(msg2, n2);
    }
    // 2) degree
    degree_kernel<<<(nE + 255) / 256, 256, 0, stream>>>(dst, deg, nE);
    // 3) scatter layer-1 messages: msg1[dst] += x[src]
    scatter_kernel<<<(nE + 255) / 256, 256, 0, stream>>>(x, src, dst, msg1, nE, D_IN);
    // 4) deg -> 1/max(deg,1)
    rdeg_kernel<<<(N + 255) / 256, 256, 0, stream>>>(deg, N);
    // 5) layer 1 fused GEMM + bias + BN + ReLU -> h
    {
        dim3 grid(D_HID / 32, (N + 127) / 128);
        sage_gemm<128, 256, true><<<grid, 256, 0, stream>>>(
            msg1, x, deg, W1l, W1r, b1, bn_g, bn_b, bn_m, bn_v, h, N);
    }
    // 6) scatter layer-2 messages: msg2[dst] += h[src]
    scatter_kernel<<<(nE + 255) / 256, 256, 0, stream>>>(h, src, dst, msg2, nE, D_HID);
    // 7) layer 2 fused GEMM + bias -> out
    {
        dim3 grid(D_OUT / 32, (N + 127) / 128);
        sage_gemm<256, 128, false><<<grid, 256, 0, stream>>>(
            msg2, h, deg, W2l, W2r, b2, nullptr, nullptr, nullptr, nullptr, out, N);
    }
}
